// CausalLocalBlock_8864812499313
// MI455X (gfx1250) — matrix operand order repaired and checked by reading, not compiled, not run on hardware
//
#include <hip/hip_runtime.h>
#include <hip/hip_bf16.h>

// ---------------- problem constants ----------------
#define BATCH 4
#define SEQ   4096
#define DIM   256
#define HID   1024
#define ROWS  (BATCH * SEQ)   // 16384
#define TAPS  15
#define LN_EPS 1e-5f

typedef __bf16 bf16_t;
typedef __attribute__((ext_vector_type(16))) __bf16 v16bf;
typedef __attribute__((ext_vector_type(8)))  float  v8f;
typedef __attribute__((ext_vector_type(4)))  int    i32x4;
typedef __attribute__((ext_vector_type(8)))  int    i32x8;
typedef __attribute__((ext_vector_type(4)))  unsigned int u32x4;

// ---------------- WMMA fragment helpers ----------------
union FragU { i32x4 q[2]; v16bf v; };

// A fragment (16x32 bf16, row-major source): lane holds row M=lane&15,
// element j -> K = (lane>>4)*16 + j. Two contiguous 16B loads.
__device__ __forceinline__ v16bf load_a_frag(const bf16_t* __restrict__ rowk, int half) {
  FragU u;
  u.q[0] = *reinterpret_cast<const i32x4*>(rowk + (half << 4));
  u.q[1] = *reinterpret_cast<const i32x4*>(rowk + 8 + (half << 4));
  return u.v;
}

__device__ __forceinline__ v16bf zero_frag() {
  FragU u;
  i32x4 z = {0, 0, 0, 0};
  u.q[0] = z; u.q[1] = z;
  return u.v;
}

// B fragment (32x16 bf16) from N-major storage: lane holds column N=lane&15,
// K = (lane>>4)*16 + j -> 16 contiguous bf16 = 32 bytes (works on LDS or global).
__device__ __forceinline__ v16bf load_b_frag(const bf16_t* p) {
  FragU u;
  u.q[0] = *reinterpret_cast<const i32x4*>(p);
  u.q[1] = *reinterpret_cast<const i32x4*>(p + 8);
  return u.v;
}

__device__ __forceinline__ v8f wmma_step(v8f c, v16bf a, v16bf b) {
  return __builtin_amdgcn_wmma_f32_16x16x32_bf16(false, a, false, b, (short)0, c, false, false);
}

__device__ __forceinline__ float gelu_exact(float x) {
  return 0.5f * x * (1.0f + erff(x * 0.70710678118654752440f));
}

// ---------------- TDM weight staging (async tensor DMA into LDS) ----------------
#if defined(__gfx1250__) && __has_builtin(__builtin_amdgcn_tensor_load_to_lds) && \
    __has_builtin(__builtin_amdgcn_s_wait_tensorcnt)
#define USE_TDM 1
#else
#define USE_TDM 0
#endif

#if USE_TDM
// 2-D tile copy: rows x (width8 * 8B) with row stride stride8 (8B units),
// packed contiguously into LDS at lds_byte_off. D# per CDNA5 ISA 8.3-8.5.
__device__ __forceinline__ void tdm_stage(const void* gsrc, unsigned lds_byte_off,
                                          unsigned width8, unsigned rows, unsigned stride8) {
  unsigned long long ga = (unsigned long long)(size_t)gsrc;
  u32x4 g0;
  g0[0] = 1u;                                            // count=1, no gather
  g0[1] = lds_byte_off;                                  // lds_addr
  g0[2] = (unsigned)(ga & 0xffffffffull);                // global_addr[31:0]
  g0[3] = (unsigned)((ga >> 32) & 0x1ffffffull)          // global_addr[56:32]
        | (2u << 30);                                    // type = 2 ("image")
  i32x8 g1;
  g1[0] = (int)(3u << 16);                               // data_size = 8B
  g1[1] = (int)((width8 & 0xffffu) << 16);               // tensor_dim0[15:0]
  g1[2] = (int)(((width8 >> 16) & 0xffffu)               // tensor_dim0[31:16]
        | ((rows & 0xffffu) << 16));                     // tensor_dim1[15:0]
  g1[3] = (int)(((rows >> 16) & 0xffffu)                 // tensor_dim1[31:16]
        | ((width8 & 0xffffu) << 16));                   // tile_dim0
  g1[4] = (int)(rows & 0xffffu);                         // tile_dim1 (tile_dim2=0)
  g1[5] = (int)stride8;                                  // tensor_dim0_stride[31:0]
  g1[6] = 0;
  g1[7] = 0;
  i32x4 z4 = {0, 0, 0, 0};
  i32x8 z8 = {0, 0, 0, 0, 0, 0, 0, 0};
  // 6-arg toolchain variant: (g0, g1, g2, g3, g4, cpol)
  __builtin_amdgcn_tensor_load_to_lds(g0, g1, z4, z4, z8, 0);
}
#define STAGE_ISSUE(gsrc, ldsptr, w8, rows, s8)                                         \
  do {                                                                                  \
    if (threadIdx.x < 32)                                                               \
      tdm_stage((const void*)(gsrc), (unsigned)(size_t)(ldsptr), (w8), (rows), (s8));   \
  } while (0)
#define STAGE_WAIT(n)                                                                   \
  do { if (threadIdx.x < 32) __builtin_amdgcn_s_wait_tensorcnt(n); } while (0)
#else
__device__ __forceinline__ void coop_stage(const bf16_t* gsrc, bf16_t* lds,
                                           unsigned width8, unsigned rows, unsigned stride8) {
  const unsigned cpr = width8 >> 1;                // 16B chunks per row
  const unsigned we = width8 * 4;                  // elements per row
  const unsigned se = stride8 * 4;                 // source stride in elements
  const unsigned chunks = rows * cpr;
  for (unsigned c = threadIdx.x; c < chunks; c += blockDim.x) {
    unsigned r = c / cpr, o = c % cpr;
    *reinterpret_cast<i32x4*>(lds + (size_t)r * we + o * 8) =
        *reinterpret_cast<const i32x4*>(gsrc + (size_t)r * se + o * 8);
  }
}
#define STAGE_ISSUE(gsrc, ldsptr, w8, rows, s8) \
  coop_stage((const bf16_t*)(gsrc), (bf16_t*)(ldsptr), (w8), (rows), (s8))
#define STAGE_WAIT(n) do { } while (0)
#endif

// ---------------- conversion kernels ----------------
__global__ void cvt_x_kernel(const float* __restrict__ in, bf16_t* __restrict__ out, size_t n) {
  size_t i = (size_t)blockIdx.x * blockDim.x + threadIdx.x;
  size_t stride = (size_t)gridDim.x * blockDim.x;
  for (; i < n; i += stride) out[i] = (bf16_t)in[i];
}

// in: [rm, cn] f32 row-major -> out: [cn, rm] bf16 (N-major for WMMA B frags)
__global__ void cvt_transpose_kernel(const float* __restrict__ in, bf16_t* __restrict__ out,
                                     int rm, int cn) {
  size_t n = (size_t)rm * cn;
  size_t i = (size_t)blockIdx.x * blockDim.x + threadIdx.x;
  size_t stride = (size_t)gridDim.x * blockDim.x;
  for (; i < n; i += stride) {
    int r = (int)(i / cn);
    int c = (int)(i % cn);
    out[(size_t)c * rm + r] = (bf16_t)in[i];
  }
}

// ---------------- conv-mix (causal 15-tap, DxD per tap) ----------------
// Block = 8 waves (8 consecutive 16-row tiles) x one 64-col strip.
// Per tap: 64x256 bf16 weight panel staged into LDS (TDM, double-buffered).
__global__ void __launch_bounds__(256) conv_mix_kernel(
    const bf16_t* __restrict__ xb, const bf16_t* __restrict__ wmixT,
    const float* __restrict__ x, const float* __restrict__ b_mix,
    float* __restrict__ y1) {
  __shared__ bf16_t sB[2][64 * DIM];   // 2 x 32 KB
  const int lane = threadIdx.x & 31;
  const int wave = threadIdx.x >> 5;
  const int half = lane >> 4;
  const int lidx = lane & 15;
  const int strip = blockIdx.x & 3;          // 4 strips of 64 cols over DIM
  const int group = blockIdx.x >> 2;         // 128 groups of 8 row tiles
  const int rt = group * 8 + wave;
  const int grow0 = rt << 4;
  const int i0 = grow0 & (SEQ - 1);
  const size_t batch_base = (size_t)(grow0 - i0) * DIM;
  const int ncol0 = (strip << 6) + lidx;

  v8f a0, a1, a2, a3;
#pragma unroll
  for (int r = 0; r < 8; ++r) { a0[r] = 0.f; a1[r] = 0.f; a2[r] = 0.f; a3[r] = 0.f; }

  const bf16_t* wbase = wmixT + (size_t)(strip << 6) * DIM;   // + k*DIM*DIM per tap
  STAGE_ISSUE(wbase, sB[0], 64, 64, 64);

  for (int k = 0; k < TAPS; ++k) {
    if (k + 1 < TAPS) {
      STAGE_ISSUE(wbase + (size_t)(k + 1) * DIM * DIM, sB[(k + 1) & 1], 64, 64, 64);
      STAGE_WAIT(1);      // oldest (tile k) complete
    } else {
      STAGE_WAIT(0);
    }
    __syncthreads();

    const bf16_t* bufk = sB[k & 1];
    const int src_i = i0 + lidx + k - (TAPS - 1);   // causal shift, zero-pad
    const bool valid = (src_i >= 0);
    const bf16_t* arow = xb + batch_base + (size_t)src_i * DIM;
#pragma unroll 2
    for (int kb = 0; kb < DIM; kb += 32) {
      v16bf af = valid ? load_a_frag(arow + kb, half) : zero_frag();
      const bf16_t* wb = bufk + kb + (half << 4);
      v16bf b0 = load_b_frag(wb + (size_t)(lidx     ) * DIM);
      v16bf b1 = load_b_frag(wb + (size_t)(lidx + 16) * DIM);
      v16bf b2 = load_b_frag(wb + (size_t)(lidx + 32) * DIM);
      v16bf b3 = load_b_frag(wb + (size_t)(lidx + 48) * DIM);
      a0 = wmma_step(a0, af, b0);
      a1 = wmma_step(a1, af, b1);
      a2 = wmma_step(a2, af, b2);
      a3 = wmma_step(a3, af, b3);
    }
    __syncthreads();   // all waves done with bufk before it is re-staged
  }

#pragma unroll
  for (int r = 0; r < 8; ++r) {
    const int row = grow0 + (half << 3) + r;
    const size_t base = (size_t)row * DIM;
    y1[base + ncol0     ] = a0[r] + b_mix[ncol0     ] + x[base + ncol0     ];
    y1[base + ncol0 + 16] = a1[r] + b_mix[ncol0 + 16] + x[base + ncol0 + 16];
    y1[base + ncol0 + 32] = a2[r] + b_mix[ncol0 + 32] + x[base + ncol0 + 32];
    y1[base + ncol0 + 48] = a3[r] + b_mix[ncol0 + 48] + x[base + ncol0 + 48];
  }
}

// ---------------- wave-per-row LayerNorm over DIM=256 ----------------
__global__ void __launch_bounds__(256) ln_kernel(
    const float* __restrict__ y, const float* __restrict__ g, const float* __restrict__ b,
    float* __restrict__ outf, bf16_t* __restrict__ outb) {
  const int row = blockIdx.x * (blockDim.x >> 5) + (threadIdx.x >> 5);
  if (row >= ROWS) return;
  const int lane = threadIdx.x & 31;
  const float* p = y + (size_t)row * DIM;
  float v[8];
  float s = 0.f;
#pragma unroll
  for (int j = 0; j < 8; ++j) { v[j] = p[lane + (j << 5)]; s += v[j]; }
#pragma unroll
  for (int o = 16; o >= 1; o >>= 1) s += __shfl_xor(s, o, 32);
  const float mu = s * (1.0f / DIM);
  float var = 0.f;
#pragma unroll
  for (int j = 0; j < 8; ++j) { float d = v[j] - mu; var += d * d; }
#pragma unroll
  for (int o = 16; o >= 1; o >>= 1) var += __shfl_xor(var, o, 32);
  const float rs = rsqrtf(var * (1.0f / DIM) + LN_EPS);
  float* po = outf + (size_t)row * DIM;
#pragma unroll
  for (int j = 0; j < 8; ++j) {
    const int c = lane + (j << 5);
    const float o_ = (v[j] - mu) * rs * g[c] + b[c];
    po[c] = o_;
    if (outb) outb[(size_t)row * DIM + c] = (bf16_t)o_;
  }
}

// ---------------- FFN1: t = gelu(h @ w_ff1 + b_ff1), bf16 out ----------------
__global__ void __launch_bounds__(256) ffn1_kernel(
    const bf16_t* __restrict__ hb, const bf16_t* __restrict__ w1T,
    const float* __restrict__ b_ff1, bf16_t* __restrict__ t) {
  __shared__ bf16_t sB[64 * DIM];     // 32 KB weight panel
  const int lane = threadIdx.x & 31;
  const int wave = threadIdx.x >> 5;
  const int half = lane >> 4;
  const int lidx = lane & 15;
  const int strip = blockIdx.x & 15;         // 16 strips of 64 cols over HID
  const int group = blockIdx.x >> 4;
  const int rt = group * 8 + wave;
  const int grow0 = rt << 4;
  const int ncol0 = (strip << 6) + lidx;
  const bf16_t* arow = hb + (size_t)(grow0 + lidx) * DIM;

  STAGE_ISSUE(w1T + (size_t)(strip << 6) * DIM, sB, 64, 64, 64);
  STAGE_WAIT(0);
  __syncthreads();

  v8f a0, a1, a2, a3;
#pragma unroll
  for (int r = 0; r < 8; ++r) { a0[r] = 0.f; a1[r] = 0.f; a2[r] = 0.f; a3[r] = 0.f; }

#pragma unroll
  for (int kb = 0; kb < DIM; kb += 32) {
    v16bf af = load_a_frag(arow + kb, half);
    const bf16_t* wb = sB + kb + (half << 4);
    v16bf b0 = load_b_frag(wb + (size_t)(lidx     ) * DIM);
    v16bf b1 = load_b_frag(wb + (size_t)(lidx + 16) * DIM);
    v16bf b2 = load_b_frag(wb + (size_t)(lidx + 32) * DIM);
    v16bf b3 = load_b_frag(wb + (size_t)(lidx + 48) * DIM);
    a0 = wmma_step(a0, af, b0);
    a1 = wmma_step(a1, af, b1);
    a2 = wmma_step(a2, af, b2);
    a3 = wmma_step(a3, af, b3);
  }

#pragma unroll
  for (int r = 0; r < 8; ++r) {
    const int row = grow0 + (half << 3) + r;
    const size_t base = (size_t)row * HID;
    t[base + ncol0     ] = (bf16_t)gelu_exact(a0[r] + b_ff1[ncol0     ]);
    t[base + ncol0 + 16] = (bf16_t)gelu_exact(a1[r] + b_ff1[ncol0 + 16]);
    t[base + ncol0 + 32] = (bf16_t)gelu_exact(a2[r] + b_ff1[ncol0 + 32]);
    t[base + ncol0 + 48] = (bf16_t)gelu_exact(a3[r] + b_ff1[ncol0 + 48]);
  }
}

// ---------------- FFN2: y2 = h + t @ w_ff2 + b_ff2 ----------------
// Reduction HID=1024 split into 4 chunks of 256; each chunk's 64x256 weight
// sub-panel (strided rows in w2T) staged via TDM, double-buffered.
__global__ void __launch_bounds__(256) ffn2_kernel(
    const bf16_t* __restrict__ t, const bf16_t* __restrict__ w2T,
    const float* __restrict__ b_ff2, const float* __restrict__ h,
    float* __restrict__ y2) {
  __shared__ bf16_t sB[2][64 * 256];  // 2 x 32 KB
  const int lane = threadIdx.x & 31;
  const int wave = threadIdx.x >> 5;
  const int half = lane >> 4;
  const int lidx = lane & 15;
  const int strip = blockIdx.x & 3;          // 4 strips of 64 cols over DIM
  const int group = blockIdx.x >> 2;
  const int rt = group * 8 + wave;
  const int grow0 = rt << 4;
  const int ncol0 = (strip << 6) + lidx;
  const bf16_t* arow = t + (size_t)(grow0 + lidx) * HID;
  const bf16_t* wbase = w2T + (size_t)(strip << 6) * HID;

  v8f a0, a1, a2, a3;
#pragma unroll
  for (int r = 0; r < 8; ++r) { a0[r] = 0.f; a1[r] = 0.f; a2[r] = 0.f; a3[r] = 0.f; }

  // rows=64 weight rows, width8 = 256 bf16 = 64 x 8B, stride8 = 1024 bf16 = 256 x 8B
  STAGE_ISSUE(wbase, sB[0], 64, 64, 256);

  for (int c = 0; c < 4; ++c) {
    if (c + 1 < 4) {
      STAGE_ISSUE(wbase + (size_t)(c + 1) * 256, sB[(c + 1) & 1], 64, 64, 256);
      STAGE_WAIT(1);
    } else {
      STAGE_WAIT(0);
    }
    __syncthreads();

    const bf16_t* bufc = sB[c & 1];
    const bf16_t* ac = arow + (size_t)c * 256;
#pragma unroll 2
    for (int kk = 0; kk < 256; kk += 32) {
      v16bf af = load_a_frag(ac + kk, half);
      const bf16_t* wb = bufc + kk + (half << 4);
      v16bf b0 = load_b_frag(wb + (size_t)(lidx     ) * 256);
      v16bf b1 = load_b_frag(wb + (size_t)(lidx + 16) * 256);
      v16bf b2 = load_b_frag(wb + (size_t)(lidx + 32) * 256);
      v16bf b3 = load_b_frag(wb + (size_t)(lidx + 48) * 256);
      a0 = wmma_step(a0, af, b0);
      a1 = wmma_step(a1, af, b1);
      a2 = wmma_step(a2, af, b2);
      a3 = wmma_step(a3, af, b3);
    }
    __syncthreads();
  }

#pragma unroll
  for (int r = 0; r < 8; ++r) {
    const int row = grow0 + (half << 3) + r;
    const size_t base = (size_t)row * DIM;
    y2[base + ncol0     ] = a0[r] + b_ff2[ncol0     ] + h[base + ncol0     ];
    y2[base + ncol0 + 16] = a1[r] + b_ff2[ncol0 + 16] + h[base + ncol0 + 16];
    y2[base + ncol0 + 32] = a2[r] + b_ff2[ncol0 + 32] + h[base + ncol0 + 32];
    y2[base + ncol0 + 48] = a3[r] + b_ff2[ncol0 + 48] + h[base + ncol0 + 48];
  }
}

// ---------------- launcher ----------------
extern "C" void kernel_launch(void* const* d_in, const int* in_sizes, int n_in,
                              void* d_out, int out_size, void* d_ws, size_t ws_size,
                              hipStream_t stream) {
  (void)in_sizes; (void)n_in; (void)out_size; (void)ws_size;
  const float* x     = (const float*)d_in[0];
  const float* w_mix = (const float*)d_in[1];
  const float* b_mix = (const float*)d_in[2];
  const float* g1    = (const float*)d_in[3];
  const float* b1    = (const float*)d_in[4];
  const float* w_ff1 = (const float*)d_in[5];
  const float* b_ff1 = (const float*)d_in[6];
  const float* w_ff2 = (const float*)d_in[7];
  const float* b_ff2 = (const float*)d_in[8];
  const float* g2    = (const float*)d_in[9];
  const float* b2    = (const float*)d_in[10];
  float* out = (float*)d_out;

  char* ws = (char*)d_ws;
  size_t off = 0;
  auto alloc = [&](size_t bytes) -> char* {
    off = (off + 255) & ~(size_t)255;
    char* p = ws + off;
    off += bytes;
    return p;
  };

  bf16_t* xb    = (bf16_t*)alloc((size_t)ROWS * DIM * sizeof(bf16_t));
  bf16_t* wmixT = (bf16_t*)alloc((size_t)TAPS * DIM * DIM * sizeof(bf16_t));
  bf16_t* w1T   = (bf16_t*)alloc((size_t)DIM * HID * sizeof(bf16_t));
  bf16_t* w2T   = (bf16_t*)alloc((size_t)HID * DIM * sizeof(bf16_t));
  float*  y1    = (float*) alloc((size_t)ROWS * DIM * sizeof(float));
  float*  hbuf  = (float*) alloc((size_t)ROWS * DIM * sizeof(float));
  bf16_t* hb    = (bf16_t*)alloc((size_t)ROWS * DIM * sizeof(bf16_t));
  bf16_t* tbuf  = (bf16_t*)alloc((size_t)ROWS * HID * sizeof(bf16_t));
  float*  y2    = (float*) alloc((size_t)ROWS * DIM * sizeof(float));

  // Stage 0: precision conversion + weight transposition (bf16, N-major)
  cvt_x_kernel<<<2048, 256, 0, stream>>>(x, xb, (size_t)ROWS * DIM);
  for (int k = 0; k < TAPS; ++k)
    cvt_transpose_kernel<<<128, 256, 0, stream>>>(w_mix + (size_t)k * DIM * DIM,
                                                  wmixT + (size_t)k * DIM * DIM, DIM, DIM);
  cvt_transpose_kernel<<<512, 256, 0, stream>>>(w_ff1, w1T, DIM, HID);
  cvt_transpose_kernel<<<512, 256, 0, stream>>>(w_ff2, w2T, HID, DIM);

  // Stage 1: causal conv mix + residual -> y1   (128 groups x 4 strips)
  conv_mix_kernel<<<128 * 4, 256, 0, stream>>>(xb, wmixT, x, b_mix, y1);

  // Stage 2: h = LN(y1)
  ln_kernel<<<ROWS / 8, 256, 0, stream>>>(y1, g1, b1, hbuf, hb);

  // Stage 3: t = gelu(h @ w_ff1 + b_ff1)       (128 groups x 16 strips)
  ffn1_kernel<<<128 * 16, 256, 0, stream>>>(hb, w1T, b_ff1, tbuf);

  // Stage 4: y2 = h + t @ w_ff2 + b_ff2        (128 groups x 4 strips)
  ffn2_kernel<<<128 * 4, 256, 0, stream>>>(tbuf, w2T, b_ff2, hbuf, y2);

  // Stage 5: out = LN(y2)
  ln_kernel<<<ROWS / 8, 256, 0, stream>>>(y2, g2, b2, out, nullptr);
}
